// GeometryOptimalTransport_47708496724567
// MI455X (gfx1250) — compile-verified, matrix operand order checked by reading, not run on hardware
//
#include <hip/hip_runtime.h>

// Problem constants (from reference setup_inputs): B=4, N=M=4096, C=128
#define B_   4
#define N_   4096
#define M_   4096
#define C_   128
#define T2_       0.04f                 // DIST_THRESH^2
#define NEG_INF_  (-1.0e9f)
#define EPS_INV_  99.99990000009999f    // 1/(EPSILON + 1e-8)

typedef __attribute__((ext_vector_type(16))) _Float16 v16h;
typedef __attribute__((ext_vector_type(8)))  float    v8f;

union ABu { unsigned u[8]; v16h h; };

static __device__ __forceinline__ unsigned pack2h(float a, float b) {
    union { _Float16 h[2]; unsigned u; } x;
    x.h[0] = (_Float16)a;
    x.h[1] = (_Float16)b;
    return x.u;
}

// ---------------------------------------------------------------------------
// init: v = 0
// ---------------------------------------------------------------------------
__global__ void ot_zero(float* __restrict__ p, int n) {
    int i = blockIdx.x * blockDim.x + threadIdx.x;
    if (i < n) p[i] = 0.0f;
}

// ---------------------------------------------------------------------------
// u[b,m] = -logsumexp_n(log_K[b,m,n] + v[b,n])
// One wave32 per target row; two-pass (max, then sum-exp) logsumexp.
// log_K recomputed on the fly from 2-D locations + masks (never materialized).
// ---------------------------------------------------------------------------
__global__ __launch_bounds__(256)
void ot_u(const float* __restrict__ slocs, const float* __restrict__ tlocs,
          const int* __restrict__ svalid, const int* __restrict__ tvalid,
          const float* __restrict__ v, float* __restrict__ u)
{
    const int lane = threadIdx.x & 31;
    const int wave = threadIdx.x >> 5;
    const int b = blockIdx.y;
    const int m = blockIdx.x * 8 + wave;

    const float2 tl = ((const float2*)tlocs)[(size_t)b * M_ + m];
    const bool   tv = tvalid[(size_t)b * M_ + m] != 0;
    const float2* sl = (const float2*)slocs + (size_t)b * N_;
    const int*    sv = svalid + (size_t)b * N_;
    const float*  vb = v + (size_t)b * N_;

    float mx = -3.0e38f;
    for (int n = lane; n < N_; n += 32) {
        float2 s = sl[n];
        float dx = tl.x - s.x, dy = tl.y - s.y;
        float d2 = dx * dx + dy * dy;
        bool ok = (d2 < T2_) && (sv[n] != 0) && tv;
        float a = (ok ? -d2 * EPS_INV_ : NEG_INF_) + vb[n];
        mx = fmaxf(mx, a);
    }
    #pragma unroll
    for (int off = 16; off > 0; off >>= 1) mx = fmaxf(mx, __shfl_xor(mx, off, 32));

    float sum = 0.0f;
    for (int n = lane; n < N_; n += 32) {
        float2 s = sl[n];
        float dx = tl.x - s.x, dy = tl.y - s.y;
        float d2 = dx * dx + dy * dy;
        bool ok = (d2 < T2_) && (sv[n] != 0) && tv;
        float a = (ok ? -d2 * EPS_INV_ : NEG_INF_) + vb[n];
        sum += __expf(a - mx);
    }
    #pragma unroll
    for (int off = 16; off > 0; off >>= 1) sum += __shfl_xor(sum, off, 32);

    if (lane == 0) u[(size_t)b * M_ + m] = -(mx + __logf(sum));
}

// ---------------------------------------------------------------------------
// v[b,n] = -logsumexp_m(log_K[b,m,n] + u[b,m]); then v = svalid ? v : 0
// ---------------------------------------------------------------------------
__global__ __launch_bounds__(256)
void ot_v(const float* __restrict__ slocs, const float* __restrict__ tlocs,
          const int* __restrict__ svalid, const int* __restrict__ tvalid,
          const float* __restrict__ u, float* __restrict__ v)
{
    const int lane = threadIdx.x & 31;
    const int wave = threadIdx.x >> 5;
    const int b = blockIdx.y;
    const int n = blockIdx.x * 8 + wave;

    const float2 sloc = ((const float2*)slocs)[(size_t)b * N_ + n];
    const bool   svn  = svalid[(size_t)b * N_ + n] != 0;
    const float2* tlv = (const float2*)tlocs + (size_t)b * M_;
    const int*    tvb = tvalid + (size_t)b * M_;
    const float*  ub  = u + (size_t)b * M_;

    float mx = -3.0e38f;
    for (int m = lane; m < M_; m += 32) {
        float2 t = tlv[m];
        float dx = t.x - sloc.x, dy = t.y - sloc.y;
        float d2 = dx * dx + dy * dy;
        bool ok = (d2 < T2_) && svn && (tvb[m] != 0);
        float a = (ok ? -d2 * EPS_INV_ : NEG_INF_) + ub[m];
        mx = fmaxf(mx, a);
    }
    #pragma unroll
    for (int off = 16; off > 0; off >>= 1) mx = fmaxf(mx, __shfl_xor(mx, off, 32));

    float sum = 0.0f;
    for (int m = lane; m < M_; m += 32) {
        float2 t = tlv[m];
        float dx = t.x - sloc.x, dy = t.y - sloc.y;
        float d2 = dx * dx + dy * dy;
        bool ok = (d2 < T2_) && svn && (tvb[m] != 0);
        float a = (ok ? -d2 * EPS_INV_ : NEG_INF_) + ub[m];
        sum += __expf(a - mx);
    }
    #pragma unroll
    for (int off = 16; off > 0; off >>= 1) sum += __shfl_xor(sum, off, 32);

    if (lane == 0) {
        float r = -(mx + __logf(sum));
        v[(size_t)b * N_ + n] = svn ? r : 0.0f;
    }
}

// ---------------------------------------------------------------------------
// Fused einsum: out[b,m,c] = sum_n attn[b,m,n] * feats[b,n,c]
// attn computed in registers (never materialized), WMMA f16 -> f32 accumulate.
// Block = 128 threads (4 waves), owns 64 rows x all 128 channels.
// Feats chunks are staged into LDS PRE-SWIZZLED into exact B-operand register
// order: per (channel-group, lane) the 8 packed-f16 dwords live as two
// 16B-aligned uint4 -> compute side is just 2x ds_load_b128 per WMMA.
// ---------------------------------------------------------------------------
__global__ __launch_bounds__(128)
void ot_einsum(const float* __restrict__ feats, const float* __restrict__ slocs,
               const float* __restrict__ tlocs, const int* __restrict__ svalid,
               const int* __restrict__ tvalid, const float* __restrict__ u,
               const float* __restrict__ v, float* __restrict__ out)
{
    __shared__ float         lv[N_];
    __shared__ float         lsx[N_];
    __shared__ float         lsy[N_];
    __shared__ unsigned char lsv[N_];
    // B-operand staging: [buf][cg][lane] -> uint4 (j=0..3) and uint4 (j=4..7)
    __shared__ uint4 lf0[2][8][32];
    __shared__ uint4 lf1[2][8][32];

    const int b   = blockIdx.y;
    const int m0  = blockIdx.x * 64;
    const int tid = threadIdx.x;
    const int lane = tid & 31, wave = tid >> 5;
    const int l0 = lane & 15, hi = lane >> 4;

    // Stage per-batch source-side tables into LDS (52 KB, reused 128x per block)
    for (int n = tid; n < N_; n += 128) {
        float2 s = ((const float2*)slocs)[(size_t)b * N_ + n];
        lsx[n] = s.x;
        lsy[n] = s.y;
        lv[n]  = v[(size_t)b * N_ + n];
        lsv[n] = (unsigned char)(svalid[(size_t)b * N_ + n] != 0);
    }

    // Per-lane A-tile row constants (A layout: lanes 0-15 and 16-31 both map M=0..15)
    const int    arow = m0 + wave * 16 + l0;
    const float2 tl   = ((const float2*)tlocs)[(size_t)b * M_ + arow];
    const float  um   = u[(size_t)b * M_ + arow];
    const bool   tv   = tvalid[(size_t)b * M_ + arow] != 0;

    auto attnv = [&](int n) -> float {
        float dx = tl.x - lsx[n], dy = tl.y - lsy[n];
        float d2 = dx * dx + dy * dy;
        bool ok = (d2 < T2_) && (lsv[n] != 0) && tv;
        // exp(log_K + u + v) * valid  (valid includes both masks)
        return ok ? __expf(um + lv[n] - d2 * EPS_INV_) : 0.0f;
    };

    v8f acc[8] = {};
    const float* fb = feats + (size_t)b * N_ * C_;

    // Staging destination constants for this thread (owns channel c = tid)
    const int scg = tid >> 4;       // channel group
    const int sl0 = tid & 15;       // column within group

    for (int n0 = 0; n0 < N_; n0 += 32) {
        const int buf = (n0 >> 5) & 1;

        // Stage feats chunk pre-swizzled: for n-pair kk, value = pack(f[2kk],f[2kk+1])
        // destination slot (j,hi) from kk(j,hi) = (j<4 ? j : j+4) + 4*hi.
        #pragma unroll
        for (int kk = 0; kk < 16; ++kk) {
            float f0 = fb[(size_t)(n0 + 2 * kk)     * C_ + tid];
            float f1 = fb[(size_t)(n0 + 2 * kk + 1) * C_ + tid];
            unsigned p = pack2h(f0, f1);
            const int shi  = (kk >> 2) & 1;                       // 0,1,0,1 per quad
            const int sj   = (kk < 8) ? (kk & 3) : ((kk & 3) + 4);
            const int slan = shi * 16 + sl0;
            if (sj < 4) ((unsigned*)&lf0[buf][scg][slan])[sj]     = p;
            else        ((unsigned*)&lf1[buf][scg][slan])[sj - 4] = p;
        }
        if (n0 + 32 < N_)
            __builtin_prefetch(&fb[(size_t)(n0 + 32) * C_ + tid], 0, 1);
        __syncthreads();

        // Build A (attn 16x32, f16): VGPR j holds K pair; low lanes K 0..7/16..23,
        // high lanes K 8..15/24..31 (CDNA5 16-bit A layout).
        ABu a;
        #pragma unroll
        for (int j = 0; j < 8; ++j) {
            const int k = ((j < 4) ? (2 * j) : (16 + 2 * (j - 4))) + 8 * hi;
            const int n = n0 + k;
            a.u[j] = pack2h(attnv(n), attnv(n + 1));
        }

        // 8 channel groups: B operand = two ds_load_b128, chained WMMA accumulate
        #pragma unroll
        for (int cg = 0; cg < 8; ++cg) {
            const uint4 q0 = lf0[buf][cg][lane];
            const uint4 q1 = lf1[buf][cg][lane];
            ABu bm;
            bm.u[0] = q0.x; bm.u[1] = q0.y; bm.u[2] = q0.z; bm.u[3] = q0.w;
            bm.u[4] = q1.x; bm.u[5] = q1.y; bm.u[6] = q1.z; bm.u[7] = q1.w;
            acc[cg] = __builtin_amdgcn_wmma_f32_16x16x32_f16(
                false, a.h, false, bm.h, (short)0, acc[cg], false, false);
        }
    }

    // Store D: VGPR i -> row (hi ? 8+i : i), col = lane (C/D 16x16 f32 layout).
    // target-mask and has_source multiplies are already folded into attn==0.
    #pragma unroll
    for (int cg = 0; cg < 8; ++cg) {
        #pragma unroll
        for (int i = 0; i < 8; ++i) {
            const int row = m0 + wave * 16 + hi * 8 + i;
            out[((size_t)b * M_ + row) * C_ + cg * 16 + l0] = acc[cg][i];
        }
    }
}

// ---------------------------------------------------------------------------
extern "C" void kernel_launch(void* const* d_in, const int* in_sizes, int n_in,
                              void* d_out, int out_size, void* d_ws, size_t ws_size,
                              hipStream_t stream) {
    (void)in_sizes; (void)n_in; (void)out_size; (void)ws_size;

    const float* feats  = (const float*)d_in[0];  // (B,N,C) f32
    const float* slocs  = (const float*)d_in[1];  // (B,N,2) f32
    const float* tlocs  = (const float*)d_in[2];  // (B,M,2) f32
    const int*   svalid = (const int*)d_in[3];    // (B,N)
    const int*   tvalid = (const int*)d_in[4];    // (B,M)
    float* out = (float*)d_out;                   // (B,M,C) f32

    float* u = (float*)d_ws;        // B*M floats
    float* v = u + (size_t)B_ * M_; // B*N floats

    ot_zero<<<dim3((B_ * N_ + 255) / 256), dim3(256), 0, stream>>>(v, B_ * N_);

    for (int it = 0; it < 3; ++it) {
        ot_u<<<dim3(M_ / 8, B_), dim3(256), 0, stream>>>(slocs, tlocs, svalid, tvalid, v, u);
        ot_v<<<dim3(N_ / 8, B_), dim3(256), 0, stream>>>(slocs, tlocs, svalid, tvalid, u, v);
    }

    ot_einsum<<<dim3(M_ / 64, B_), dim3(128), 0, stream>>>(
        feats, slocs, tlocs, svalid, tvalid, u, v, out);
}
